// POD_GalerkinTransformer_75505525064766
// MI455X (gfx1250) — matrix operand order repaired and checked by reading, not compiled, not run on hardware
//
#include <hip/hip_runtime.h>
#include <math.h>

// ---------- types ----------
typedef __bf16 bf16;
typedef bf16  v8bf  __attribute__((ext_vector_type(8)));
typedef bf16  v16bf __attribute__((ext_vector_type(16)));
typedef float v8f   __attribute__((ext_vector_type(8)));

#define LDSK 40   // padded LDS K-stride (halves): 80B rows, 16B-aligned sublanes

// ---------- WMMA helper ----------
__device__ __forceinline__ v8f wmma_bf16(v16bf a, v16bf b, v8f c) {
  return __builtin_amdgcn_wmma_f32_16x16x32_bf16(false, a, false, b, (short)0, c, false, false);
}

// Load a 16-element bf16 fragment from LDS laid out [row][LDSK].
__device__ __forceinline__ v16bf ldfrag(const bf16* base, int row, int klo, int khi) {
  v8bf lo = *(const v8bf*)(base + row * LDSK + klo);
  v8bf hi = *(const v8bf*)(base + row * LDSK + khi);
  return __builtin_shufflevector(lo, hi, 0,1,2,3,4,5,6,7,8,9,10,11,12,13,14,15);
}

// ---------- transpose+convert: W[K][N] fp32 -> Wt[N][K] bf16 (one-time pass) ----------
__global__ __launch_bounds__(256) void cvt_t_kernel(const float* __restrict__ s,
                                                    bf16* __restrict__ d, int K, int N) {
  __shared__ bf16 tile[32][33];
  int n0 = blockIdx.x * 32, k0 = blockIdx.y * 32;
  const float* sp = s + (size_t)blockIdx.z * K * N;
  bf16* dp = d + (size_t)blockIdx.z * K * N;
  int tx = threadIdx.x & 31, ty = threadIdx.x >> 5;  // 32 x 8
  #pragma unroll
  for (int i = 0; i < 32; i += 8)
    tile[ty + i][tx] = (bf16)sp[(size_t)(k0 + ty + i) * N + n0 + tx];
  __syncthreads();
  #pragma unroll
  for (int i = 0; i < 32; i += 8)
    dp[(size_t)(n0 + ty + i) * K + k0 + tx] = tile[tx][ty + i];
}

// ---------- pack x[b,t,n,c] -> A[b*n, t*2+c] bf16 ----------
__global__ __launch_bounds__(256) void pack_x_kernel(const float* __restrict__ x,
                                                     bf16* __restrict__ ap) {
  size_t idx = (size_t)blockIdx.x * blockDim.x + threadIdx.x;  // b*n*t
  if (idx >= (size_t)4 * 4096 * 64) return;
  int t = (int)(idx & 63);
  size_t bn = idx >> 6;
  int n = (int)(bn & 4095);
  int b = (int)(bn >> 12);
  float2 v = *(const float2*)(x + (((size_t)(b * 64 + t) * 4096) + n) * 2);
  bf16* o = ap + bn * 128 + t * 2;
  o[0] = (bf16)v.x;
  o[1] = (bf16)v.y;
}

// ---------- layernorm over 512, fp32 in -> bf16 out ----------
__global__ __launch_bounds__(256) void ln_kernel(const float* __restrict__ h,
                                                 const float* __restrict__ g,
                                                 const float* __restrict__ bta,
                                                 bf16* __restrict__ y) {
  int row = blockIdx.x;
  int tid = threadIdx.x;
  const float* hr = h + (size_t)row * 512;
  float2 xv = *(const float2*)(hr + tid * 2);
  float s = xv.x + xv.y;
  float s2 = xv.x * xv.x + xv.y * xv.y;
  #pragma unroll
  for (int m = 16; m; m >>= 1) { s += __shfl_xor(s, m); s2 += __shfl_xor(s2, m); }
  __shared__ float ps[8], ps2[8];
  int w = tid >> 5, lane = tid & 31;
  if (lane == 0) { ps[w] = s; ps2[w] = s2; }
  __syncthreads();
  if (tid == 0) {
    float a = 0.f, b = 0.f;
    #pragma unroll
    for (int i = 0; i < 8; ++i) { a += ps[i]; b += ps2[i]; }
    ps[0] = a; ps2[0] = b;
  }
  __syncthreads();
  float mean = ps[0] * (1.f / 512.f);
  float var  = ps2[0] * (1.f / 512.f) - mean * mean;
  float inv  = rsqrtf(var + 1e-5f);
  int c0 = tid * 2;
  bf16* yr = y + (size_t)row * 512 + c0;
  yr[0] = (bf16)((xv.x - mean) * inv * g[c0]     + bta[c0]);
  yr[1] = (bf16)((xv.y - mean) * inv * g[c0 + 1] + bta[c0 + 1]);
}

// ---------- instance-norm of k and v (per row, per head, over d=64), in-place bf16 ----------
__global__ __launch_bounds__(256) void inorm_kernel(bf16* __restrict__ qkv) {
  int row = blockIdx.x;
  int w = threadIdx.x >> 5;
  int lane = threadIdx.x & 31;
  bf16* base = qkv + (size_t)row * 1536 + 512 + w * 64;
  #pragma unroll
  for (int part = 0; part < 2; ++part) {
    bf16* p = base + part * 512;
    float a = (float)p[lane], b = (float)p[lane + 32];
    float s = a + b, s2 = a * a + b * b;
    #pragma unroll
    for (int m = 16; m; m >>= 1) { s += __shfl_xor(s, m); s2 += __shfl_xor(s2, m); }
    float mean = s * (1.f / 64.f);
    float var  = s2 * (1.f / 64.f) - mean * mean;
    float inv  = rsqrtf(var + 1e-5f);
    p[lane]      = (bf16)((a - mean) * inv);
    p[lane + 32] = (bf16)((b - mean) * inv);
  }
}

// ---------- dotsT[bh][e][d] = (k^T v)^T / n per (b,h), reduction over n=4096 ----------
__global__ __launch_bounds__(256) void dots_kernel(const bf16* __restrict__ qkv,
                                                   bf16* __restrict__ dotsT) {
  int bh = blockIdx.x;
  int b = bh >> 3, hh = bh & 7;
  __shared__ __align__(16) bf16 Ks[64 * LDSK];
  __shared__ __align__(16) bf16 Vs[64 * LDSK];
  int tid = threadIdx.x, w = tid >> 5, lane = tid & 31;
  int dm = w & 3, en0 = (w >> 2) * 2;
  v8f acc[2] = {{}, {}};
  size_t rowBase = (size_t)b * 4096;
  int aklo = (lane & 16) ? 16 : 0;
  int bklo = (lane & 16) ? 16 : 0;
  int li = tid >> 3, d0 = (tid & 7) * 8;
  for (int nc = 0; nc < 4096; nc += 32) {
    const bf16* src = qkv + (rowBase + nc + li) * 1536 + 512 + hh * 64 + d0;
    v8bf kv = *(const v8bf*)src;
    v8bf vv = *(const v8bf*)(src + 512);
    #pragma unroll
    for (int j = 0; j < 8; ++j) {
      Ks[(d0 + j) * LDSK + li] = kv[j];
      Vs[(d0 + j) * LDSK + li] = vv[j];
    }
    __syncthreads();
    v16bf af = ldfrag(Ks, dm * 16 + (lane & 15), aklo, aklo + 8);
    #pragma unroll
    for (int tn = 0; tn < 2; ++tn) {
      v16bf bfq = ldfrag(Vs, (en0 + tn) * 16 + (lane & 15), bklo, bklo + 8);
      acc[tn] = wmma_bf16(af, bfq, acc[tn]);
    }
    __syncthreads();
  }
  const float inv_n = 1.0f / 4096.0f;
  #pragma unroll
  for (int tn = 0; tn < 2; ++tn) {
    int e = (en0 + tn) * 16 + (lane & 15);
    int dr0 = dm * 16 + ((lane & 16) ? 8 : 0);
    #pragma unroll
    for (int r = 0; r < 8; ++r)   // transposed store: [e][d]
      dotsT[(size_t)bh * 4096 + (size_t)e * 64 + dr0 + r] = (bf16)(acc[tn][r] * inv_n);
  }
}

// ---------- generic tiled bf16 GEMM, 128xNT tile, K-step 32, double-buffered LDS ----------
// B operand is pre-transposed: Bt[N][K], row stride ldb.
// MODE 0: embed   : Cf = acc*sqrt(512) + bias[col] + pos_enc(n, col)
// MODE 1: bf16    : Cbf = bf16(acc)
// MODE 3: resid+y : t = Cf + acc + bias; Cf = t; Cbf = bf16(t)
// MODE 4: decode  : out[b, col, n] = acc + bias[col]
template <int MODE, int NT>
__global__ __launch_bounds__(256) void gemm_bf16(
    const bf16* __restrict__ A, const bf16* __restrict__ Bt,
    int lda, int ldb, int K, int Z2,
    long long aS1, long long aS2, long long bS1, long long bS2,
    long long cS1, long long cS2,
    const float* __restrict__ bias,
    float* __restrict__ Cf, int ldcf,
    bf16* __restrict__ Cbf, int ldcb) {
  constexpr int NW = NT / 2;    // per-wave N extent
  constexpr int TN = NW / 16;   // B tiles per wave
  __shared__ __align__(16) bf16 As[2][128 * LDSK];
  __shared__ __align__(16) bf16 Bs[2][NT * LDSK];
  int n0 = blockIdx.x * NT, m0 = blockIdx.y * 128;
  int z = blockIdx.z;
  int z1 = z / Z2, z2 = z - z1 * Z2;
  long long aoff = (long long)z1 * aS1 + (long long)z2 * aS2;
  long long boff = (long long)z1 * bS1 + (long long)z2 * bS2;
  long long coff = (long long)z1 * cS1 + (long long)z2 * cS2;
  int tid = threadIdx.x, w = tid >> 5, lane = tid & 31;
  int wm = w & 3, wn = w >> 2;  // 4 waves along M, 2 along N
  v8f acc[2][TN];
  {
    v8f zz = {};
    #pragma unroll
    for (int a = 0; a < 2; ++a)
      #pragma unroll
      for (int b = 0; b < TN; ++b) acc[a][b] = zz;
  }
  int aklo = (lane & 16) ? 16 : 0;
  int bklo = (lane & 16) ? 16 : 0;
  int ra = tid >> 1, ha = (tid & 1) * 16;                       // A stage map
  int rb = (NT == 128) ? (tid >> 1) : (tid >> 2);               // B stage map
  int kb = (NT == 128) ? ((tid & 1) * 16) : ((tid & 3) * 8);
  v8bf ga0, ga1, gb0, gb1;
  auto loadA = [&](int k0) {
    const bf16* s = A + aoff + (size_t)(m0 + ra) * lda + k0 + ha;
    ga0 = *(const v8bf*)s;
    ga1 = *(const v8bf*)(s + 8);
  };
  auto loadB = [&](int k0) {
    const bf16* s = Bt + boff + (size_t)(n0 + rb) * ldb + k0 + kb;
    gb0 = *(const v8bf*)s;
    if (NT == 128) gb1 = *(const v8bf*)(s + 8);
  };
  auto storeA = [&](int bi) {
    *(v8bf*)(As[bi] + ra * LDSK + ha)     = ga0;
    *(v8bf*)(As[bi] + ra * LDSK + ha + 8) = ga1;
  };
  auto storeB = [&](int bi) {
    *(v8bf*)(Bs[bi] + rb * LDSK + kb) = gb0;
    if (NT == 128) *(v8bf*)(Bs[bi] + rb * LDSK + kb + 8) = gb1;
  };
  loadA(0); loadB(0);
  storeA(0); storeB(0);
  __syncthreads();
  int buf = 0;
  for (int k0 = 0; k0 < K; k0 += 32) {
    bool more = (k0 + 32) < K;
    if (more) { loadA(k0 + 32); loadB(k0 + 32); }   // global -> regs, overlapped with WMMA
    v16bf af[2], bfq[TN];
    #pragma unroll
    for (int tm = 0; tm < 2; ++tm)
      af[tm] = ldfrag(As[buf], wm * 32 + tm * 16 + (lane & 15), aklo, aklo + 8);
    #pragma unroll
    for (int tn = 0; tn < TN; ++tn)
      bfq[tn] = ldfrag(Bs[buf], wn * NW + tn * 16 + (lane & 15), bklo, bklo + 8);
    #pragma unroll
    for (int tm = 0; tm < 2; ++tm)
      #pragma unroll
      for (int tn = 0; tn < TN; ++tn)
        acc[tm][tn] = wmma_bf16(af[tm], bfq[tn], acc[tm][tn]);
    if (more) { storeA(buf ^ 1); storeB(buf ^ 1); } // regs -> other LDS buffer
    __syncthreads();
    buf ^= 1;
  }
  // epilogue
  #pragma unroll
  for (int tm = 0; tm < 2; ++tm) {
    #pragma unroll
    for (int tn = 0; tn < TN; ++tn) {
      int col = n0 + wn * NW + tn * 16 + (lane & 15);
      int rbase = m0 + wm * 32 + tm * 16 + ((lane & 16) ? 8 : 0);
      #pragma unroll
      for (int r = 0; r < 8; ++r) {
        int row = rbase + r;
        float v = acc[tm][tn][r];
        if (MODE == 0) {
          int n = row & 4095;
          int j2 = col & ~1;
          float arg = (float)n * __expf(-(float)j2 * (9.210340371976184f / 512.0f));
          float pe = (col & 1) ? __cosf(arg) : __sinf(arg);
          Cf[(size_t)row * ldcf + col] = v * 22.627416998f + bias[col] + pe;
        } else if (MODE == 1) {
          Cbf[coff + (size_t)row * ldcb + col] = (bf16)v;
        } else if (MODE == 3) {
          float* p = Cf + (size_t)row * ldcf + col;
          float t = *p + v + bias[col];
          *p = t;
          Cbf[(size_t)row * ldcb + col] = (bf16)t;
        } else if (MODE == 4) {
          int bb = row >> 12, n = row & 4095;
          Cf[((size_t)bb * 128 + col) * 4096 + n] = v + bias[col];
        }
      }
    }
  }
}

// ---------- dual-B GEMM with fused geglu (W1 pre-transposed: W1t[4096][512]) ----------
__global__ __launch_bounds__(256) void gemm_geglu(const bf16* __restrict__ A,
                                                  const bf16* __restrict__ B1t,
                                                  const float* __restrict__ bias,
                                                  bf16* __restrict__ G) {
  __shared__ __align__(16) bf16 As[2][128 * LDSK];
  __shared__ __align__(16) bf16 B0s[2][64 * LDSK];
  __shared__ __align__(16) bf16 B1s[2][64 * LDSK];
  int n0 = blockIdx.x * 64, m0 = blockIdx.y * 128;
  int tid = threadIdx.x, w = tid >> 5, lane = tid & 31;
  int wm = w & 3, wn = w >> 2;
  v8f accA[2][2], accB[2][2];
  {
    v8f zz = {};
    #pragma unroll
    for (int a = 0; a < 2; ++a)
      #pragma unroll
      for (int b = 0; b < 2; ++b) { accA[a][b] = zz; accB[a][b] = zz; }
  }
  int aklo = (lane & 16) ? 16 : 0;
  int bklo = (lane & 16) ? 16 : 0;
  int ra = tid >> 1, ha = (tid & 1) * 16;
  int rb = tid >> 2, kb = (tid & 3) * 8;
  v8bf ga0, ga1, g0, g1;
  auto loadAB = [&](int k0) {
    const bf16* s = A + (size_t)(m0 + ra) * 512 + k0 + ha;
    ga0 = *(const v8bf*)s;
    ga1 = *(const v8bf*)(s + 8);
    const bf16* t = B1t + (size_t)(n0 + rb) * 512 + k0 + kb;
    g0 = *(const v8bf*)t;                         // W1a^T rows [n0, n0+64)
    g1 = *(const v8bf*)(t + (size_t)2048 * 512);  // W1b^T rows [2048+n0, ...)
  };
  auto storeAB = [&](int bi) {
    *(v8bf*)(As[bi] + ra * LDSK + ha)     = ga0;
    *(v8bf*)(As[bi] + ra * LDSK + ha + 8) = ga1;
    *(v8bf*)(B0s[bi] + rb * LDSK + kb) = g0;
    *(v8bf*)(B1s[bi] + rb * LDSK + kb) = g1;
  };
  loadAB(0);
  storeAB(0);
  __syncthreads();
  int buf = 0;
  for (int k0 = 0; k0 < 512; k0 += 32) {
    bool more = (k0 + 32) < 512;
    if (more) loadAB(k0 + 32);
    v16bf af[2];
    #pragma unroll
    for (int tm = 0; tm < 2; ++tm)
      af[tm] = ldfrag(As[buf], wm * 32 + tm * 16 + (lane & 15), aklo, aklo + 8);
    #pragma unroll
    for (int tn = 0; tn < 2; ++tn) {
      v16bf b0 = ldfrag(B0s[buf], wn * 32 + tn * 16 + (lane & 15), bklo, bklo + 8);
      v16bf b1 = ldfrag(B1s[buf], wn * 32 + tn * 16 + (lane & 15), bklo, bklo + 8);
      #pragma unroll
      for (int tm = 0; tm < 2; ++tm) {
        accA[tm][tn] = wmma_bf16(af[tm], b0, accA[tm][tn]);
        accB[tm][tn] = wmma_bf16(af[tm], b1, accB[tm][tn]);
      }
    }
    if (more) storeAB(buf ^ 1);
    __syncthreads();
    buf ^= 1;
  }
  #pragma unroll
  for (int tm = 0; tm < 2; ++tm) {
    #pragma unroll
    for (int tn = 0; tn < 2; ++tn) {
      int col = n0 + wn * 32 + tn * 16 + (lane & 15);
      float ba = bias[col], bb = bias[col + 2048];
      int rbase = m0 + wm * 32 + tm * 16 + ((lane & 16) ? 8 : 0);
      #pragma unroll
      for (int r = 0; r < 8; ++r) {
        float xa = accA[tm][tn][r] + ba;
        float xb = accB[tm][tn][r] + bb;
        float ge = 0.5f * xa * (1.0f + erff(xa * 0.70710678118f));
        G[(size_t)(rbase + r) * 2048 + col] = (bf16)(ge * xb);
      }
    }
  }
}

// ---------- host launch ----------
extern "C" void kernel_launch(void* const* d_in, const int* in_sizes, int n_in,
                              void* d_out, int out_size, void* d_ws, size_t ws_size,
                              hipStream_t stream) {
  const float* x     = (const float*)d_in[0];
  const float* W_emb = (const float*)d_in[1];
  const float* b_emb = (const float*)d_in[2];
  const float* ln_g  = (const float*)d_in[3];
  const float* ln_b  = (const float*)d_in[4];
  const float* Wqkv  = (const float*)d_in[5];
  const float* Wout  = (const float*)d_in[6];
  const float* b_out = (const float*)d_in[7];
  const float* W1    = (const float*)d_in[8];
  const float* b1    = (const float*)d_in[9];
  const float* W2    = (const float*)d_in[10];
  const float* b2    = (const float*)d_in[11];
  const float* Wd    = (const float*)d_in[12];
  const float* bd    = (const float*)d_in[13];

  char* wsp = (char*)d_ws;
  auto alloc = [&](size_t bytes) -> void* {
    void* p = (void*)wsp;
    wsp += (bytes + 255) & ~(size_t)255;
    return p;
  };
  bf16* wEmbT = (bf16*)alloc((size_t)128 * 512 * 2);
  bf16* wQkvT = (bf16*)alloc((size_t)6 * 512 * 1536 * 2);
  bf16* wOutT = (bf16*)alloc((size_t)6 * 512 * 512 * 2);
  bf16* w1T   = (bf16*)alloc((size_t)6 * 512 * 4096 * 2);
  bf16* w2T   = (bf16*)alloc((size_t)6 * 2048 * 512 * 2);
  bf16* wDT   = (bf16*)alloc((size_t)512 * 128 * 2);
  bf16* apack = (bf16*)alloc((size_t)16384 * 128 * 2);
  float* h    = (float*)alloc((size_t)16384 * 512 * 4);
  bf16* y     = (bf16*)alloc((size_t)16384 * 512 * 2);
  bf16* y2    = (bf16*)alloc((size_t)16384 * 512 * 2);
  bf16* qkv   = (bf16*)alloc((size_t)16384 * 1536 * 2);
  bf16* dotsT = (bf16*)alloc((size_t)32 * 4096 * 2);
  bf16* g     = (bf16*)alloc((size_t)16384 * 2048 * 2);

  auto cvtT = [&](const float* s, bf16* d, int K, int N, int L) {
    cvt_t_kernel<<<dim3(N / 32, K / 32, L), 256, 0, stream>>>(s, d, K, N);
  };
  cvtT(W_emb, wEmbT, 128, 512, 1);
  cvtT(Wqkv,  wQkvT, 512, 1536, 6);
  cvtT(Wout,  wOutT, 512, 512, 6);
  cvtT(W1,    w1T,   512, 4096, 6);
  cvtT(W2,    w2T,   2048, 512, 6);
  cvtT(Wd,    wDT,   512, 128, 1);

  pack_x_kernel<<<(unsigned)(((size_t)4 * 4096 * 64 + 255) / 256), 256, 0, stream>>>(x, apack);

  // embed: h = (apack @ W_emb) * sqrt(512) + b_emb + pos_enc
  gemm_bf16<0, 128><<<dim3(4, 128, 1), 256, 0, stream>>>(
      apack, wEmbT, 128, 128, 128, 1, 0, 0, 0, 0, 0, 0,
      b_emb, h, 512, nullptr, 0);

  for (int i = 0; i < 6; ++i) {
    // y = layernorm(h)
    ln_kernel<<<16384, 256, 0, stream>>>(h, ln_g + i * 512, ln_b + i * 512, y);
    // qkv = y @ Wqkv[i]
    gemm_bf16<1, 128><<<dim3(12, 128, 1), 256, 0, stream>>>(
        y, wQkvT + (size_t)i * 512 * 1536, 512, 512, 512, 1,
        0, 0, 0, 0, 0, 0, nullptr, nullptr, 0, qkv, 1536);
    // k,v instance-norm (in place)
    inorm_kernel<<<16384, 256, 0, stream>>>(qkv);
    // dotsT[bh] = ((k^T v)/n)^T
    dots_kernel<<<32, 256, 0, stream>>>(qkv, dotsT);
    // o = q @ dots   (batched over b=4, h=8; write bf16 into y)
    gemm_bf16<1, 64><<<dim3(1, 32, 32), 256, 0, stream>>>(
        qkv, dotsT, 1536, 64, 64, 8,
        (long long)4096 * 1536, 64, (long long)8 * 4096, 4096,
        (long long)4096 * 512, 64, nullptr, nullptr, 0, y, 512);
    // h += o @ Wout[i] + b_out[i];  y2 = bf16(h)
    gemm_bf16<3, 128><<<dim3(4, 128, 1), 256, 0, stream>>>(
        y, wOutT + (size_t)i * 512 * 512, 512, 512, 512, 1,
        0, 0, 0, 0, 0, 0, b_out + i * 512, h, 512, y2, 512);
    // g = gelu(y2@W1a + b1a) * (y2@W1b + b1b)
    gemm_geglu<<<dim3(32, 128, 1), 256, 0, stream>>>(
        y2, w1T + (size_t)i * 512 * 4096, b1 + i * 4096, g);
    // h += g @ W2[i] + b2[i];  y2 = bf16(h)
    gemm_bf16<3, 128><<<dim3(4, 128, 1), 256, 0, stream>>>(
        g, w2T + (size_t)i * 2048 * 512, 2048, 2048, 2048, 1,
        0, 0, 0, 0, 0, 0, b2 + i * 512, h, 512, y2, 512);
  }

  // decode: out[b, col, n] = y2 @ Wd + bd  (transposed scatter)
  gemm_bf16<4, 128><<<dim3(1, 128, 1), 256, 0, stream>>>(
      y2, wDT, 512, 512, 512, 1,
      0, 0, 0, 0, 0, 0, bd, (float*)d_out, 0, nullptr, 0);
}
